// EdgeUpdate_38628935860831
// MI455X (gfx1250) — matrix operand order repaired and checked by reading, not compiled, not run on hardware
//
#include <hip/hip_runtime.h>
#include <stdint.h>

#define N_NODES_C  100000
#define N_EDGES_C  1000000
#define HIDDEN_C   128
#define EDGE_DIM_C 64
#define IN_DIM_C   320      // 2*HIDDEN + EDGE_DIM
#define BM         128      // edges per block
#define NTHREADS   512      // 16 waves: 8 row-groups x 2 column-halves

// LDS row strides in DWORDs (16 consecutive rows -> distinct bank groups,
// 16-byte fragment loads stay 16B-aligned: stride % 4 == 0)
#define SIN_STRIDE 164      // 328 bf16 (320 + 8 pad)
#define SW1_STRIDE 164
#define SH_STRIDE  68       // 136 bf16 (128 + 8 pad)
#define SW2_STRIDE 68

typedef __attribute__((ext_vector_type(16))) __bf16 v16bf;
typedef __attribute__((ext_vector_type(8)))  float  v8f;
typedef __attribute__((ext_vector_type(4)))  float  f32x4;
typedef __attribute__((ext_vector_type(2)))  float  f32x2;

union Frag {            // one WMMA A/B operand: 32 bytes = 8 VGPRs
    v16bf v;
    f32x4 f[2];
};

__device__ __forceinline__ unsigned short f32_to_bf16(float f) {
    uint32_t u = __builtin_bit_cast(uint32_t, f);
    uint32_t r = u + 0x7FFFu + ((u >> 16) & 1u);   // round-to-nearest-even
    return (unsigned short)(r >> 16);
}

__device__ __forceinline__ uint32_t pack_bf16x2(float lo, float hi) {
    return (uint32_t)f32_to_bf16(lo) | ((uint32_t)f32_to_bf16(hi) << 16);
}

// gfx1250 async global->LDS copy (16B per lane). VDST = per-lane LDS byte
// address, VADDR = 64-bit global address, GV mode (saddr = off).
__device__ __forceinline__ void async_b128_to_lds(uint32_t lds_addr, const void* gaddr) {
    asm volatile("global_load_async_to_lds_b128 %0, %1, off"
                 :: "v"(lds_addr), "v"(gaddr) : "memory");
}
__device__ __forceinline__ void wait_asynccnt0() {
    asm volatile("s_wait_asynccnt 0x0" ::: "memory");
}

// For a __shared__ object, the generic (flat) pointer keeps the allocation-
// relative LDS byte offset in bits [31:0] (LDS aperture), which is exactly
// what the async-copy VDST operand expects.
__device__ __forceinline__ uint32_t lds_offset(const void* p) {
    return (uint32_t)(uintptr_t)p;
}

// ---------------------------------------------------------------------------
// Prep: W1 [320][128] f32 -> W1t [128][320] bf16 ; W2 [128][128] -> W2t [128][128] bf16
// ---------------------------------------------------------------------------
__global__ __launch_bounds__(256) void prep_weights_kernel(
    const float* __restrict__ W1, const float* __restrict__ W2,
    unsigned short* __restrict__ gW1t, unsigned short* __restrict__ gW2t)
{
    int idx = blockIdx.x * 256 + threadIdx.x;
    if (idx < IN_DIM_C * HIDDEN_C) {
        int n = idx / IN_DIM_C;
        int k = idx - n * IN_DIM_C;
        gW1t[idx] = f32_to_bf16(W1[k * HIDDEN_C + n]);
    }
    if (idx < HIDDEN_C * HIDDEN_C) {
        int n = idx >> 7;
        int k = idx & 127;
        gW2t[idx] = f32_to_bf16(W2[k * HIDDEN_C + n]);
    }
}

// ---------------------------------------------------------------------------
// Main fused kernel: async weight staging || gather -> GEMM1 -> GELU -> GEMM2
// ---------------------------------------------------------------------------
__global__ __launch_bounds__(NTHREADS) void edge_mlp_kernel(
    const float* __restrict__ x,
    const int*   __restrict__ edge_index,   // [2][E]
    const float* __restrict__ edge_attr,
    const float* __restrict__ b1,
    const float* __restrict__ b2,
    const unsigned short* __restrict__ gW1t,  // [128][320] bf16
    const unsigned short* __restrict__ gW2t,  // [128][128] bf16
    float* __restrict__ out)
{
    __shared__ uint32_t sIn[BM * SIN_STRIDE];      // input tile, bf16 pairs
    __shared__ uint32_t sW1[HIDDEN_C * SW1_STRIDE];
    __shared__ uint32_t sW2[HIDDEN_C * SW2_STRIDE];
    __shared__ uint32_t sH [BM * SH_STRIDE];       // hidden activations, bf16 pairs

    const int tid  = threadIdx.x;
    const int w    = tid >> 5;        // wave id 0..15
    const int lane = tid & 31;
    const int lh   = lane >> 4;       // lane half
    const int lm   = lane & 15;
    const int rg   = w >> 1;          // row-group 0..7
    const int cg   = w & 1;           // column half 0..1
    const int m0   = rg * 16;         // this wave's row base within the tile
    const long blockStart = (long)blockIdx.x * BM;

    // ---- kick off async W1t/W2t (bf16) -> LDS copies; they run during gather ----
    {
        // W1t: 128 rows x 40 x b128 chunks = 5120 chunks
        for (int c = tid; c < HIDDEN_C * 40; c += NTHREADS) {
            int n = c / 40, j = c - n * 40;                  // j: 16B chunk in row
            uint32_t dst = lds_offset(&sW1[n * SW1_STRIDE + j * 4]);
            async_b128_to_lds(dst, gW1t + (size_t)n * 320 + j * 8);
        }
        // W2t: 128 rows x 16 x b128 chunks = 2048 chunks
        for (int c = tid; c < HIDDEN_C * 16; c += NTHREADS) {
            int n = c >> 4, j = c & 15;
            uint32_t dst = lds_offset(&sW2[n * SW2_STRIDE + j * 4]);
            async_b128_to_lds(dst, gW2t + (size_t)n * 128 + j * 8);
        }
    }

    // ---- gather + f32->bf16 convert the [128 x 320] input tile; one wave per row ----
    for (int p = 0; p < 8; ++p) {
        int row = p * 16 + w;
        long e  = blockStart + row;
        long ec = e < N_EDGES_C ? e : (N_EDGES_C - 1);
        int src = edge_index[ec];
        int dst = edge_index[N_EDGES_C + ec];
        f32x4 xs = *(const f32x4*)(x + (long)src * HIDDEN_C + lane * 4);
        f32x4 xd = *(const f32x4*)(x + (long)dst * HIDDEN_C + lane * 4);
        f32x2 ea = *(const f32x2*)(edge_attr + ec * EDGE_DIM_C + lane * 2);
        uint32_t* r = sIn + row * SIN_STRIDE;
        r[lane * 2]          = pack_bf16x2(xs.x, xs.y);
        r[lane * 2 + 1]      = pack_bf16x2(xs.z, xs.w);
        r[64 + lane * 2]     = pack_bf16x2(xd.x, xd.y);
        r[64 + lane * 2 + 1] = pack_bf16x2(xd.z, xd.w);
        r[128 + lane]        = pack_bf16x2(ea.x, ea.y);
    }
    wait_asynccnt0();          // weight copies landed in LDS
    __syncthreads();

    const v8f vzero = {0.f, 0.f, 0.f, 0.f, 0.f, 0.f, 0.f, 0.f};

    // ================= Layer 1: [16 x 320] x [320 x 64] per wave =================
    v8f acc[4];
#pragma unroll
    for (int t = 0; t < 4; ++t) acc[t] = vzero;

    {
        const uint32_t* aRow = sIn + (m0 + lm) * SIN_STRIDE;
        for (int k0 = 0; k0 < IN_DIM_C; k0 += 32) {
            Frag a;
            int ad = (k0 >> 1) + 4 * lh;              // A: lanes0-15 K 0-7/16-23, lanes16-31 K 8-15/24-31
            a.f[0] = *(const f32x4*)(aRow + ad);
            a.f[1] = *(const f32x4*)(aRow + ad + 8);
            int bd = (k0 >> 1) + 4 * lh;              // B: lane = N, lanes0-15 K 0-7/16-23, lanes16-31 K 8-15/24-31
#pragma unroll
            for (int t = 0; t < 4; ++t) {
                Frag b;
                const uint32_t* bRow = sW1 + ((cg * 4 + t) * 16 + lm) * SW1_STRIDE + bd;
                b.f[0] = *(const f32x4*)(bRow);
                b.f[1] = *(const f32x4*)(bRow + 8);
                acc[t] = __builtin_amdgcn_wmma_f32_16x16x32_bf16(
                    false, a.v, false, b.v, (short)0, acc[t], false, false);
            }
        }
    }

    // ---- bias + exact GELU -> bf16 into sH ----
    {
        unsigned short* hp = (unsigned short*)sH;
#pragma unroll
        for (int t = 0; t < 4; ++t) {
            float bias = b1[(cg * 4 + t) * 16 + lm];
#pragma unroll
            for (int r = 0; r < 8; ++r) {
                float v = acc[t][r] + bias;
                v = 0.5f * v * (1.0f + erff(v * 0.70710678118654752f));
                int m = m0 + r + 8 * lh;
                int n = (cg * 4 + t) * 16 + lm;
                hp[m * (SH_STRIDE * 2) + n] = f32_to_bf16(v);
            }
        }
    }
    __syncthreads();   // layer-2 A rows now mix both column halves

    // ================= Layer 2: [16 x 128] x [128 x 64] per wave =================
    v8f acc2[4];
#pragma unroll
    for (int t = 0; t < 4; ++t) acc2[t] = vzero;

    {
        const uint32_t* hRow = sH + (m0 + lm) * SH_STRIDE;
#pragma unroll
        for (int k0 = 0; k0 < HIDDEN_C; k0 += 32) {
            Frag a;
            int ad = (k0 >> 1) + 4 * lh;
            a.f[0] = *(const f32x4*)(hRow + ad);
            a.f[1] = *(const f32x4*)(hRow + ad + 8);
            int bd = (k0 >> 1) + 4 * lh;
#pragma unroll
            for (int t = 0; t < 4; ++t) {
                Frag b;
                const uint32_t* bRow = sW2 + ((cg * 4 + t) * 16 + lm) * SW2_STRIDE + bd;
                b.f[0] = *(const f32x4*)(bRow);
                b.f[1] = *(const f32x4*)(bRow + 8);
                acc2[t] = __builtin_amdgcn_wmma_f32_16x16x32_bf16(
                    false, a.v, false, b.v, (short)0, acc2[t], false, false);
            }
        }
    }

    // ---- bias + store f32 output ----
#pragma unroll
    for (int t = 0; t < 4; ++t) {
        float bias = b2[(cg * 4 + t) * 16 + lm];
#pragma unroll
        for (int r = 0; r < 8; ++r) {
            long e = blockStart + m0 + r + 8 * lh;
            if (e < N_EDGES_C) {
                out[e * HIDDEN_C + (cg * 4 + t) * 16 + lm] = acc2[t][r] + bias;
            }
        }
    }
}

// ---------------------------------------------------------------------------
extern "C" void kernel_launch(void* const* d_in, const int* in_sizes, int n_in,
                              void* d_out, int out_size, void* d_ws, size_t ws_size,
                              hipStream_t stream) {
    (void)in_sizes; (void)n_in; (void)out_size; (void)ws_size;
    const float* x   = (const float*)d_in[0];
    const int*   ei  = (const int*)d_in[1];      // per harness: integer -> const int*
    const float* ea  = (const float*)d_in[2];
    const float* W1  = (const float*)d_in[3];
    const float* b1  = (const float*)d_in[4];
    const float* W2  = (const float*)d_in[5];
    const float* b2  = (const float*)d_in[6];
    float*       out = (float*)d_out;

    unsigned short* gW1t = (unsigned short*)d_ws;                 // 320*128 bf16
    unsigned short* gW2t = gW1t + IN_DIM_C * HIDDEN_C;            // 128*128 bf16

    prep_weights_kernel<<<(IN_DIM_C * HIDDEN_C + 255) / 256, 256, 0, stream>>>(
        W1, W2, gW1t, gW2t);

    int nblk = (N_EDGES_C + BM - 1) / BM;
    edge_mlp_kernel<<<nblk, NTHREADS, 0, stream>>>(x, ei, ea, b1, b2, gW1t, gW2t, out);
}
